// MultiHeadAttention_7980049236527
// MI455X (gfx1250) — matrix operand order repaired and checked by reading, not compiled, not run on hardware
//
#include <hip/hip_runtime.h>
#include <hip/hip_bf16.h>

#define S_  2048
#define B_  2
#define D_  512
#define H_  8
#define HD_ 64

typedef __attribute__((ext_vector_type(16))) _Float16 v16h;
typedef __attribute__((ext_vector_type(8)))  _Float16 v8h;
typedef __attribute__((ext_vector_type(8)))  float    v8f;

static __device__ __forceinline__ v8f wmma_f16(v16h a, v16h b, v8f c) {
  // D = A(16x32 f16) * B(32x16 f16) + C(16x16 f32)
  return __builtin_amdgcn_wmma_f32_16x16x32_f16(false, a, false, b, (short)0, c, false, false);
}

// ---- fragment loaders (contiguous k order k = 16*(lane/16) + s, wave32) ---------------------

// A 16x32 f16 from row-major f32 (convert). src = tile origin (row0,k0), 32B-aligned rows.
static __device__ __forceinline__ v16h ld_a_f32(const float* src, int lda, int lane) {
  int m  = lane & 15;
  int kb = (lane >> 4) << 4;          // lanes 0-15: K 0-15 ; 16-31: K 16-31
  const float* p = src + (size_t)m * lda + kb;
  v8f lo = *(const v8f*)p;
  v8f hi = *(const v8f*)(p + 8);
  v16h a;
#pragma unroll
  for (int i = 0; i < 8; ++i) { a[i] = (_Float16)lo[i]; a[8 + i] = (_Float16)hi[i]; }
  return a;
}

// A 16x32 from row-major f16 (global or LDS); per-lane data = two 16B chunks.
static __device__ __forceinline__ v16h ld_a_f16(const _Float16* src, int lda, int lane) {
  int m  = lane & 15;
  int kb = (lane >> 4) << 4;
  const _Float16* p = src + (size_t)m * lda + kb;
  v8h lo = *(const v8h*)p;
  v8h hi = *(const v8h*)(p + 8);
  v16h a;
#pragma unroll
  for (int i = 0; i < 8; ++i) { a[i] = lo[i]; a[8 + i] = hi[i]; }
  return a;
}

// B 32x16 f16, per-lane 16 contiguous K-halfs (32B vector load).
// B[k][n] = src[n*stride_n + k]; lanes 0-15: N=lane,K 0-15; 16-31: K 16-31.
static __device__ __forceinline__ v16h ld_b_kcontig_f16(const _Float16* src, int stride_n, int lane) {
  int n  = lane & 15;
  int kb = (lane >> 4) << 4;
  return *(const v16h*)(src + (size_t)n * stride_n + kb);
}

// B 32x16 from row-major f32 B[k][n] = src[k*stride_k + n] (strided, convert).
static __device__ __forceinline__ v16h ld_b_kstride_f32(const float* src, int stride_k, int lane) {
  int n  = lane & 15;
  int kb = (lane >> 4) << 4;
  const float* p = src + (size_t)kb * stride_k + n;
  v16h b;
#pragma unroll
  for (int i = 0; i < 16; ++i) b[i] = (_Float16)p[(size_t)i * stride_k];
  return b;
}

// ---- CDNA5 async global -> LDS (ASYNCcnt) ----------------------------------
static __device__ __forceinline__ uint32_t lds_addr32(const void* p) {
  return (uint32_t)(uintptr_t)p;       // generic shared ptr: addr[31:0] == LDS offset
}
static __device__ __forceinline__ void async_copy_b128(uint32_t lds_off, const void* gaddr) {
  asm volatile("global_load_async_to_lds_b128 %0, %1, off"
               :: "v"(lds_off), "v"(gaddr) : "memory");
}

// ---------------------------------------------------------------------------
// Kernel 1: hoisted rel-pos bias gather: bias[s,t] = pos_bias[idx0[s,t], idx1[s,t]]
// (broadcast over b,h in the reference; 16MB map stays L2-resident)
// ---------------------------------------------------------------------------
__global__ void bias_gather_kernel(const float* __restrict__ pos_bias,
                                   const int* __restrict__ idx0,
                                   const int* __restrict__ idx1,
                                   float* __restrict__ bias) {
  size_t i = (size_t)blockIdx.x * blockDim.x + threadIdx.x;
  if (i < (size_t)S_ * S_)
    bias[i] = pos_bias[(size_t)idx0[i] * (2 * S_ - 1) + idx1[i]];
}

// ---------------------------------------------------------------------------
// Kernel 2: Q/K/V projections (WMMA GEMM, one wave per 16x16 tile).
// q,k -> f16 [B,H,S,HD]; v -> TRANSPOSED f16 [B,H,HD,S].
// ---------------------------------------------------------------------------
__global__ __launch_bounds__(32) void proj_kernel(
    const float* __restrict__ q_in, const float* __restrict__ k_in, const float* __restrict__ v_in,
    const float* __restrict__ Wq, const float* __restrict__ bq,
    const float* __restrict__ Wk, const float* __restrict__ bk,
    const float* __restrict__ Wv, const float* __restrict__ bv,
    _Float16* __restrict__ qf, _Float16* __restrict__ kf, _Float16* __restrict__ vtf) {
  int lane  = threadIdx.x;
  int n0    = blockIdx.x * 16;     // output col (h*HD+hd)
  int row0  = blockIdx.y * 16;     // output row (b*S+s)
  int which = blockIdx.z;          // 0=q 1=k 2=v
  const float* X  = which == 0 ? q_in : (which == 1 ? k_in : v_in);
  const float* W  = which == 0 ? Wq   : (which == 1 ? Wk   : Wv);
  const float* bb = which == 0 ? bq   : (which == 1 ? bk   : bv);

  v8f acc = {};
  for (int k0 = 0; k0 < D_; k0 += 32) {
    v16h a = ld_a_f32(X + (size_t)row0 * D_ + k0, D_, lane);
    // weight as B[k=d][n=h*HD+hd] = W[(h*D + d)*HD + hd]
    int n = n0 + (lane & 15);
    int kb = k0 + ((lane >> 4) << 4);
    int h = n / HD_, hd = n % HD_;
    const float* wp = W + ((size_t)h * D_ + kb) * HD_ + hd;
    v16h b;
#pragma unroll
    for (int i = 0; i < 16; ++i) b[i] = (_Float16)wp[(size_t)i * HD_];
    acc = wmma_f16(a, b, acc);
  }

  int n = n0 + (lane & 15);
  int h = n / HD_, hd = n % HD_;
  float bn = bb[n];
  int half = lane >> 4;
#pragma unroll
  for (int j = 0; j < 8; ++j) {
    int row = row0 + j + 8 * half;  // C layout: VGPR j -> M = j + 8*(lane>=16)
    int bidx = row / S_, s = row % S_;
    float val = acc[j] + bn;
    if (which == 2)
      vtf[(((size_t)bidx * H_ + h) * HD_ + hd) * S_ + s] = (_Float16)val;
    else
      (which == 0 ? qf : kf)[(((size_t)bidx * H_ + h) * S_ + s) * HD_ + hd] = (_Float16)val;
  }
}

// ---------------------------------------------------------------------------
// Kernel 3: flash attention, one wave per (b,h,16 query rows).
// K/V tiles double-buffered in LDS via global_load_async_to_lds_b128
// (ASYNCcnt pipeline, depth 2). P transposed C->A via LDS.
// ---------------------------------------------------------------------------

// Stage one 32-wide K/V tile (16 async b128 ops): K tile = contiguous 4KB,
// V (transposed) tile = 64 rows x 64B.
static __device__ __forceinline__ void stage_tile(const _Float16* kh, const _Float16* vh,
                                                  int t0, uint32_t koff, uint32_t voff, int lane) {
  const char* kg = (const char*)(kh + (size_t)t0 * HD_);
#pragma unroll
  for (int i = 0; i < 8; ++i) {
    uint32_t o = (uint32_t)(lane * 16 + i * 512);
    async_copy_b128(koff + o, kg + o);
  }
#pragma unroll
  for (int i = 0; i < 8; ++i) {
    int chunk = i * 32 + lane;                 // 0..255 chunks of 16B
    int row  = chunk >> 2;                     // hd row
    int coff = (chunk & 3) * 16;               // byte offset within 64B row
    const char* vg = (const char*)(vh + (size_t)row * S_ + t0) + coff;
    async_copy_b128(voff + (uint32_t)chunk * 16, vg);
  }
}

__global__ __launch_bounds__(32) void attn_kernel(
    const _Float16* __restrict__ qf, const _Float16* __restrict__ kf,
    const _Float16* __restrict__ vtf, const float* __restrict__ bias,
    _Float16* __restrict__ rep) {
  __shared__ _Float16 lds_k[2][32 * HD_];   // [t_local][hd]  4KB x2
  __shared__ _Float16 lds_v[2][HD_ * 32];   // [hd][t_local]  4KB x2
  __shared__ _Float16 lds_p[16 * 32];       // P tile, row-major

  int lane = threadIdx.x;
  int s0 = blockIdx.x * 16;
  int h  = blockIdx.y;
  int b  = blockIdx.z;
  int half = lane >> 4, col = lane & 15;

  const _Float16* qh = qf  + ((size_t)b * H_ + h) * S_ * HD_;
  const _Float16* kh = kf  + ((size_t)b * H_ + h) * S_ * HD_;
  const _Float16* vh = vtf + ((size_t)b * H_ + h) * HD_ * S_;

  v16h qa0 = ld_a_f16(qh + (size_t)s0 * HD_ + 0,  HD_, lane);   // K dim = hd 0..31
  v16h qa1 = ld_a_f16(qh + (size_t)s0 * HD_ + 32, HD_, lane);   // hd 32..63

  v8f o0 = {}, o1 = {}, o2 = {}, o3 = {};
  float m_r[8], l_r[8];
#pragma unroll
  for (int j = 0; j < 8; ++j) { m_r[j] = -1e30f; l_r[j] = 0.f; }

  int nblk = (s0 + 16 + 31) / 32;           // causal: key blocks with t <= s0+15
  uint32_t koff[2] = { lds_addr32(&lds_k[0][0]), lds_addr32(&lds_k[1][0]) };
  uint32_t voff[2] = { lds_addr32(&lds_v[0][0]), lds_addr32(&lds_v[1][0]) };

  stage_tile(kh, vh, 0, koff[0], voff[0], lane);          // prologue: tile 0

  for (int tb = 0; tb < nblk; ++tb) {
    int t0 = tb * 32;
    int cur = tb & 1;
    if (tb + 1 < nblk) {
      stage_tile(kh, vh, t0 + 32, koff[(tb + 1) & 1], voff[(tb + 1) & 1], lane);
      asm volatile("s_wait_asynccnt 0x10" ::: "memory");  // tile tb fully in LDS
    } else {
      asm volatile("s_wait_asynccnt 0x0" ::: "memory");
    }
    const _Float16* lk = &lds_k[cur][0];
    const _Float16* lv = &lds_v[cur][0];

    // ---- scores: two 16x16 tiles covering t in [t0, t0+32) ----
    v8f sc[2];
#pragma unroll
    for (int tt = 0; tt < 2; ++tt) {
      v16h kb0 = ld_b_kcontig_f16(lk + (size_t)(tt * 16) * HD_ + 0,  HD_, lane);
      v16h kb1 = ld_b_kcontig_f16(lk + (size_t)(tt * 16) * HD_ + 32, HD_, lane);
      v8f c = {};
      c = wmma_f16(qa0, kb0, c);
      c = wmma_f16(qa1, kb1, c);
      sc[tt] = c;
    }
    // ---- scale + rel-pos bias + causal mask ----
#pragma unroll
    for (int tt = 0; tt < 2; ++tt) {
      int tg = t0 + tt * 16 + col;
#pragma unroll
      for (int j = 0; j < 8; ++j) {
        int sg = s0 + j + 8 * half;
        float v = sc[tt][j] * 0.125f + bias[(size_t)sg * S_ + tg];
        sc[tt][j] = (tg > sg) ? -1e30f : v;
      }
    }
    // ---- online softmax (row reductions across 16-lane half-wave) ----
    float alpha[8];
#pragma unroll
    for (int j = 0; j < 8; ++j) {
      float rm = fmaxf(sc[0][j], sc[1][j]);
#pragma unroll
      for (int i = 1; i < 16; i <<= 1) rm = fmaxf(rm, __shfl_xor(rm, i, 32));
      float mn = fmaxf(m_r[j], rm);
      alpha[j] = __expf(m_r[j] - mn);
      m_r[j] = mn;
      float p0 = __expf(sc[0][j] - mn);
      float p1 = __expf(sc[1][j] - mn);
      sc[0][j] = p0; sc[1][j] = p1;
      float rs = p0 + p1;
#pragma unroll
      for (int i = 1; i < 16; i <<= 1) rs += __shfl_xor(rs, i, 32);
      l_r[j] = l_r[j] * alpha[j] + rs;
    }
#pragma unroll
    for (int j = 0; j < 8; ++j) {
      o0[j] *= alpha[j]; o1[j] *= alpha[j]; o2[j] *= alpha[j]; o3[j] *= alpha[j];
    }
    // ---- P tile: C layout -> row-major LDS -> A fragment ----
#pragma unroll
    for (int j = 0; j < 8; ++j) {
      int row = j + 8 * half;
      lds_p[row * 32 + col]      = (_Float16)sc[0][j];
      lds_p[row * 32 + 16 + col] = (_Float16)sc[1][j];
    }
    asm volatile("s_wait_dscnt 0" ::: "memory");   // same-wave LDS RAW
    __syncthreads();
    v16h pa = ld_a_f16(lds_p, 32, lane);           // A 16x32 (K = t-local)
    // ---- O += P @ V (V tile in LDS, [hd][t_local], K contiguous) ----
    v16h vb0 = ld_b_kcontig_f16(lv + (size_t)(0 * 16) * 32, 32, lane);
    v16h vb1 = ld_b_kcontig_f16(lv + (size_t)(1 * 16) * 32, 32, lane);
    v16h vb2 = ld_b_kcontig_f16(lv + (size_t)(2 * 16) * 32, 32, lane);
    v16h vb3 = ld_b_kcontig_f16(lv + (size_t)(3 * 16) * 32, 32, lane);
    o0 = wmma_f16(pa, vb0, o0);
    o1 = wmma_f16(pa, vb1, o1);
    o2 = wmma_f16(pa, vb2, o2);
    o3 = wmma_f16(pa, vb3, o3);
    __syncthreads();
  }
  // ---- epilogue: normalize and store concatenated heads (f16 rep) ----
#pragma unroll
  for (int j = 0; j < 8; ++j) {
    int sg = s0 + j + 8 * half;
    float inv = 1.0f / l_r[j];
    size_t base = ((size_t)b * S_ + sg) * D_ + (size_t)h * HD_;
    rep[base + 0 * 16 + col] = (_Float16)(o0[j] * inv);
    rep[base + 1 * 16 + col] = (_Float16)(o1[j] * inv);
    rep[base + 2 * 16 + col] = (_Float16)(o2[j] * inv);
    rep[base + 3 * 16 + col] = (_Float16)(o3[j] * inv);
  }
}

// ---------------------------------------------------------------------------
// Kernel 4: output projection  out = rep @ Wo + bo  (f32 result)
// ---------------------------------------------------------------------------
__global__ __launch_bounds__(32) void out_kernel(
    const _Float16* __restrict__ rep, const float* __restrict__ Wo,
    const float* __restrict__ bo, float* __restrict__ out) {
  int lane = threadIdx.x;
  int n0   = blockIdx.x * 16;
  int row0 = blockIdx.y * 16;
  v8f acc = {};
  for (int k0 = 0; k0 < D_; k0 += 32) {
    v16h a = ld_a_f16(rep + (size_t)row0 * D_ + k0, D_, lane);
    v16h b = ld_b_kstride_f32(Wo + (size_t)k0 * D_ + n0, D_, lane);
    acc = wmma_f16(a, b, acc);
  }
  int n = n0 + (lane & 15);
  float bn = bo[n];
  int half = lane >> 4;
#pragma unroll
  for (int j = 0; j < 8; ++j) {
    int row = row0 + j + 8 * half;
    out[(size_t)row * D_ + n] = acc[j] + bn;
  }
}

// ---------------------------------------------------------------------------
extern "C" void kernel_launch(void* const* d_in, const int* in_sizes, int n_in,
                              void* d_out, int out_size, void* d_ws, size_t ws_size,
                              hipStream_t stream) {
  const float* query = (const float*)d_in[0];
  const float* key   = (const float*)d_in[1];
  const float* value = (const float*)d_in[2];
  const float* Wq = (const float*)d_in[3];
  const float* bq = (const float*)d_in[4];
  const float* Wk = (const float*)d_in[5];
  const float* bk = (const float*)d_in[6];
  const float* Wv = (const float*)d_in[7];
  const float* bv = (const float*)d_in[8];
  const float* Wo = (const float*)d_in[9];
  const float* bo = (const float*)d_in[10];
  const float* pos_bias = (const float*)d_in[11];
  const int* idx0 = (const int*)d_in[12];
  const int* idx1 = (const int*)d_in[13];
  // d_in[14] = causal triu mask, applied analytically (t > s)

  // workspace layout (32 MB total)
  char* ws = (char*)d_ws;
  float*    bias = (float*)ws;                                   // S*S f32   = 16 MB
  _Float16* qf   = (_Float16*)(ws + (size_t)S_ * S_ * 4);        // B,H,S,HD  =  4 MB
  _Float16* kf   = qf  + (size_t)B_ * H_ * S_ * HD_;             //           =  4 MB
  _Float16* vtf  = kf  + (size_t)B_ * H_ * S_ * HD_;             // B,H,HD,S  =  4 MB
  _Float16* rep  = vtf + (size_t)B_ * H_ * S_ * HD_;             // B*S, D    =  4 MB

  bias_gather_kernel<<<dim3(((size_t)S_ * S_ + 255) / 256), 256, 0, stream>>>(
      pos_bias, idx0, idx1, bias);

  proj_kernel<<<dim3(D_ / 16, (B_ * S_) / 16, 3), 32, 0, stream>>>(
      query, key, value, Wq, bq, Wk, bk, Wv, bv, qf, kf, vtf);

  attn_kernel<<<dim3(S_ / 16, H_, B_), 32, 0, stream>>>(qf, kf, vtf, bias, rep);

  out_kernel<<<dim3(D_ / 16, (B_ * S_) / 16), 32, 0, stream>>>(rep, Wo, bo, (float*)d_out);
}
